// ProtoAttention_65335042507425
// MI455X (gfx1250) — matrix operand order repaired and checked by reading, not compiled, not run on hardware
//
#include <hip/hip_runtime.h>

// ProtoAttention fused flash kernel for gfx1250 (MI455X).
// out[b] = softmax(x[b] @ protos^T + gumbel(u)) @ protos
// B=16, N=2048, D=1024, P=2048, tau=1.

#define BB 16
#define NN 2048
#define DD 1024
#define PP 2048

typedef __attribute__((ext_vector_type(16))) __bf16 bf16x16;
typedef __attribute__((ext_vector_type(8)))  float  f32x8;

union BFrag {
    bf16x16 bf;
    uint4 q[2];
    unsigned short s[16];
};

__device__ __forceinline__ unsigned short f2bf(float f) {
    unsigned int x = __float_as_uint(f);
    x += 0x7FFFu + ((x >> 16) & 1u);      // round-to-nearest-even
    return (unsigned short)(x >> 16);
}
__device__ __forceinline__ unsigned int pack2bf(float a, float b) {
    return (unsigned int)f2bf(a) | ((unsigned int)f2bf(b) << 16);
}

// Pre-pass: protos f32 -> bf16 (4 MiB) so the fused kernel can async-DMA it.
__global__ __launch_bounds__(256)
void convert_protos(const float* __restrict__ src, unsigned short* __restrict__ dst) {
    int i = (blockIdx.x * 256 + threadIdx.x) * 4;
    float4 v = *(const float4*)(src + i);
    *(uint2*)(dst + i) = make_uint2(pack2bf(v.x, v.y), pack2bf(v.z, v.w));
}

__global__ __launch_bounds__(256, 1)
void proto_attn_fused(const float* __restrict__ x,
                      const float* __restrict__ u,
                      const float* __restrict__ protos,
                      const unsigned short* __restrict__ protos_bf,
                      float* __restrict__ out,
                      int use_async)
{
    // LDS (~215 KB of the 320 KB WGP budget)
    __shared__ __align__(16) unsigned short x_tile[32 * DD];   // [n][d] bf16, 64KB
    __shared__ __align__(16) unsigned short p_tile[32 * DD];   // [p][d] bf16, 64KB (GEMM1 B)
    __shared__ __align__(16) unsigned short pT_tile[DD * 32];  // [d][p] bf16, 64KB (GEMM2 B)
    __shared__ float s_red[2][4][16][32];                      // split-K partial S, 16KB
    __shared__ __align__(16) unsigned short pw[2][16 * 32];    // softmax probs bf16 [n][p]
    __shared__ float corr[2][16];                              // per-row rescale factors
    __shared__ float lsum[2][16];                              // per-row softmax denominators

    const int tid  = threadIdx.x;
    const int lane = tid & 31;
    const int w    = tid >> 5;   // wave 0..7
    const int wn   = w & 1;      // n-group: rows [wn*16, wn*16+16)
    const int wd   = w >> 1;     // d-slice: cols [wd*256, wd*256+256)
    const int l15  = lane & 15;
    const int lh   = lane >> 4;  // lane half (WMMA M/K split)

    const int b  = blockIdx.y;
    const int n0 = blockIdx.x * 32;

    // ---- stage x row-tile once: f32 -> bf16 into LDS ----
    {
        const float* xs = x + ((size_t)b * NN + n0) * DD;
        #pragma unroll
        for (int i = 0; i < 32; ++i) {
            int c  = tid + 256 * i;      // 8192 chunks of 4 floats
            int fi = c * 4;
            float4 v = *(const float4*)(xs + fi);
            *(uint2*)(&x_tile[fi]) = make_uint2(pack2bf(v.x, v.y), pack2bf(v.z, v.w));
        }
    }

    f32x8 acc[16];
    #pragma unroll
    for (int t = 0; t < 16; ++t)
        #pragma unroll
        for (int r = 0; r < 8; ++r)
            acc[t][r] = 0.0f;

    float m_run = __int_as_float(0xff800000);  // -inf
    float l_run = 0.0f;

    // LDS byte offset of p_tile for async-DMA destination addressing
    const unsigned p_tile_lds = (unsigned)(uintptr_t)&p_tile[0];

    for (int p0 = 0; p0 < PP; p0 += 32) {
        __syncthreads();   // protect p_tile/pT_tile/pw/s_red reuse

        if (use_async) {
            // ---- async DMA: bf16 protos tile straight into LDS (row-major) ----
            // 64KB tile = 4096 x 16B chunks; each thread issues 16 async b128 copies.
            const char* gbase = (const char*)(protos_bf + (size_t)p0 * DD);
            #pragma unroll
            for (int i = 0; i < 16; ++i) {
                int c = tid + 256 * i;
                unsigned laddr = p_tile_lds + c * 16;
                unsigned long long ga = (unsigned long long)(gbase) + (unsigned long long)(c * 16);
                asm volatile("global_load_async_to_lds_b128 %0, %1, off"
                             :: "v"(laddr), "v"(ga) : "memory");
            }
            // ---- transposed copy from bf16 protos (gather along p, pack, ds store) ----
            #pragma unroll
            for (int i = 0; i < 32; ++i) {
                int c = tid + 256 * i;       // [8 p-groups][1024 d]
                int d = c & 1023;
                int p = (c >> 10) * 4;
                const unsigned short* pb = protos_bf + (size_t)(p0 + p) * DD + d;
                unsigned v0 = pb[0 * DD];
                unsigned v1 = pb[1 * DD];
                unsigned v2 = pb[2 * DD];
                unsigned v3 = pb[3 * DD];
                *(uint2*)(&pT_tile[d * 32 + p]) =
                    make_uint2(v0 | (v1 << 16), v2 | (v3 << 16));
            }
            asm volatile("s_wait_asynccnt 0" ::: "memory");
        } else {
            // ---- fallback: stage from f32 protos with on-the-fly conversion ----
            const float* ps = protos + (size_t)p0 * DD;
            #pragma unroll
            for (int i = 0; i < 32; ++i) {
                int c  = tid + 256 * i;
                int fi = c * 4;
                float4 v = *(const float4*)(ps + fi);
                *(uint2*)(&p_tile[fi]) = make_uint2(pack2bf(v.x, v.y), pack2bf(v.z, v.w));
            }
            #pragma unroll
            for (int i = 0; i < 32; ++i) {
                int c = tid + 256 * i;       // [8 p-groups][1024 d]
                int d = c & 1023;
                int p = (c >> 10) * 4;
                float v0 = ps[(p + 0) * DD + d];
                float v1 = ps[(p + 1) * DD + d];
                float v2 = ps[(p + 2) * DD + d];
                float v3 = ps[(p + 3) * DD + d];
                *(uint2*)(&pT_tile[d * 32 + p]) =
                    make_uint2(pack2bf(v0, v1), pack2bf(v2, v3));
            }
        }
        __syncthreads();

        // ---- gumbel prefetch for softmax-owner lanes (overlaps GEMM1) ----
        float g[32];
        if (wd == 0 && lane < 16) {
            int n = n0 + wn * 16 + lane;
            #pragma unroll
            for (int k = 0; k < 32; ++k) {
                float uu = u[((size_t)b * PP + p0 + k) * NN + n];
                g[k] = -__logf(-__logf(uu));
            }
        }

        // ---- GEMM1: partial S[16n x 32p] over d-slice [wd*256, wd*256+256) ----
        f32x8 s0, s1;
        #pragma unroll
        for (int r = 0; r < 8; ++r) { s0[r] = 0.0f; s1[r] = 0.0f; }

        #pragma unroll
        for (int ks = 0; ks < 8; ++ks) {
            const int d0 = wd * 256 + ks * 32;
            // A frag (x): lane row n=l15; elems 0..15 = K d0+(lh?16:0)+0..15
            BFrag a, b0f, b1f;
            const unsigned short* ar = &x_tile[(wn * 16 + l15) * DD + d0 + lh * 16];
            a.q[0] = *(const uint4*)(ar);
            a.q[1] = *(const uint4*)(ar + 8);
            // B frags (protos, K=d contiguous): col p = l15 (+16 for second subtile)
            const unsigned short* br0 = &p_tile[l15 * DD + d0 + lh * 16];
            b0f.q[0] = *(const uint4*)(br0);
            b0f.q[1] = *(const uint4*)(br0 + 8);
            const unsigned short* br1 = &p_tile[(16 + l15) * DD + d0 + lh * 16];
            b1f.q[0] = *(const uint4*)(br1);
            b1f.q[1] = *(const uint4*)(br1 + 8);

            s0 = __builtin_amdgcn_wmma_f32_16x16x32_bf16(false, a.bf, false, b0f.bf,
                                                         (short)0, s0, false, false);
            s1 = __builtin_amdgcn_wmma_f32_16x16x32_bf16(false, a.bf, false, b1f.bf,
                                                         (short)0, s1, false, false);
        }
        // write split-K partials (C layout: lane<16 -> M=r, lane>=16 -> M=r+8)
        #pragma unroll
        for (int r = 0; r < 8; ++r) {
            int row = lh * 8 + r;
            s_red[wn][wd][row][l15]      = s0[r];
            s_red[wn][wd][row][16 + l15] = s1[r];
        }
        __syncthreads();

        // ---- online softmax over this p-tile (one wave per n-group, 16 lanes) ----
        if (wd == 0 && lane < 16) {
            float s[32];
            #pragma unroll
            for (int k = 0; k < 32; ++k) {
                s[k] = s_red[wn][0][lane][k] + s_red[wn][1][lane][k]
                     + s_red[wn][2][lane][k] + s_red[wn][3][lane][k] + g[k];
            }
            float tmax = s[0];
            #pragma unroll
            for (int k = 1; k < 32; ++k) tmax = fmaxf(tmax, s[k]);
            float mn = fmaxf(m_run, tmax);
            float cf = __expf(m_run - mn);
            float ssum = 0.0f;
            #pragma unroll
            for (int k = 0; k < 32; ++k) { s[k] = __expf(s[k] - mn); ssum += s[k]; }
            l_run = l_run * cf + ssum;
            m_run = mn;
            corr[wn][lane] = cf;
            unsigned int* pr = (unsigned int*)&pw[wn][lane * 32];
            #pragma unroll
            for (int e = 0; e < 16; ++e) pr[e] = pack2bf(s[2 * e], s[2 * e + 1]);
        }
        __syncthreads();

        // ---- GEMM2: acc[16n x 256d] = acc*corr + Pw[16x32] @ protos[32p x 256d] ----
        float cf8[8];
        #pragma unroll
        for (int r = 0; r < 8; ++r) cf8[r] = corr[wn][lh * 8 + r];

        BFrag pa;  // A frag = probabilities, row n=l15, K=p
        const unsigned short* pwr = &pw[wn][l15 * 32 + lh * 16];
        pa.q[0] = *(const uint4*)(pwr);
        pa.q[1] = *(const uint4*)(pwr + 8);

        #pragma unroll
        for (int t = 0; t < 16; ++t) {
            const int dcol = wd * 256 + t * 16 + l15;
            BFrag pb;  // B frag from transposed tile: col=d, K=p contiguous
            const unsigned short* btr = &pT_tile[dcol * 32 + lh * 16];
            pb.q[0] = *(const uint4*)(btr);
            pb.q[1] = *(const uint4*)(btr + 8);

            #pragma unroll
            for (int r = 0; r < 8; ++r) acc[t][r] *= cf8[r];

            acc[t] = __builtin_amdgcn_wmma_f32_16x16x32_bf16(false, pa.bf, false, pb.bf,
                                                             (short)0, acc[t], false, false);
        }
    }

    // ---- final normalization by softmax denominator and store ----
    if (wd == 0 && lane < 16) lsum[wn][lane] = l_run;
    __syncthreads();

    float inv[8];
    #pragma unroll
    for (int r = 0; r < 8; ++r) inv[r] = 1.0f / lsum[wn][lh * 8 + r];

    float* ob = out + ((size_t)b * NN + n0 + wn * 16) * DD;
    #pragma unroll
    for (int t = 0; t < 16; ++t) {
        int dcol = wd * 256 + t * 16 + l15;
        #pragma unroll
        for (int r = 0; r < 8; ++r) {
            int row = lh * 8 + r;
            ob[(size_t)row * DD + dcol] = acc[t][r] * inv[r];
        }
    }
}

extern "C" void kernel_launch(void* const* d_in, const int* in_sizes, int n_in,
                              void* d_out, int out_size, void* d_ws, size_t ws_size,
                              hipStream_t stream) {
    const float* x      = (const float*)d_in[0];
    const float* u      = (const float*)d_in[1];
    const float* protos = (const float*)d_in[2];
    float* out          = (float*)d_out;

    const size_t protos_bf_bytes = (size_t)PP * DD * sizeof(unsigned short);
    int use_async = (d_ws != nullptr && ws_size >= protos_bf_bytes) ? 1 : 0;
    unsigned short* protos_bf = (unsigned short*)d_ws;

    if (use_async) {
        // 2M elements, 4 per thread -> 2048 blocks of 256
        convert_protos<<<dim3((PP * DD) / (256 * 4)), 256, 0, stream>>>(protos, protos_bf);
    } else {
        protos_bf = (unsigned short*)protos;  // unused branch, keep pointer valid
    }

    dim3 grid(NN / 32, BB);
    proto_attn_fused<<<grid, 256, 0, stream>>>(x, u, protos, protos_bf, out, use_async);

    (void)in_sizes; (void)n_in; (void)out_size;
}
